// MPNN_44736379355551
// MI455X (gfx1250) — compile-verified, matrix operand order checked by reading, not run on hardware
//
#include <hip/hip_runtime.h>

// ---------------------------------------------------------------------------
// CDNA5 (gfx1250) MPNN: bf16 WMMA for node linear + edge MLP, f32 accumulate.
// Edge gather uses GLOBAL_LOAD_ASYNC_TO_LDS_B128 (ASYNCcnt) on gfx1250.
// ---------------------------------------------------------------------------

typedef __attribute__((ext_vector_type(16))) __bf16    v16bf;
typedef __attribute__((ext_vector_type(2)))  __bf16    v2bf;
typedef __attribute__((ext_vector_type(8)))  float     v8f;
typedef __attribute__((ext_vector_type(4)))  unsigned  v4u;

#if __has_builtin(__builtin_amdgcn_cvt_pk_bf16_f32)
__device__ __forceinline__ unsigned pack_bf16(float lo, float hi) {
    union { v2bf v; unsigned u; } c;
    c.v = __builtin_amdgcn_cvt_pk_bf16_f32(lo, hi);
    return c.u;
}
#else
__device__ __forceinline__ unsigned pack_bf16(float lo, float hi) {
    unsigned ul = __float_as_uint(lo);
    ul += 0x7FFFu + ((ul >> 16) & 1u);
    unsigned uh = __float_as_uint(hi);
    uh += 0x7FFFu + ((uh >> 16) & 1u);
    return (ul >> 16) | (uh & 0xFFFF0000u);
}
#endif
__device__ __forceinline__ unsigned short f32_to_bf16(float f) {
    return (unsigned short)(pack_bf16(f, 0.0f) & 0xFFFFu);
}
// K offset of packed pair j (0..7) inside a 16x16x32 bf16 fragment, per lane half.
__device__ __forceinline__ int frag_k0(int j, int khi) {
    return ((j < 4) ? (2 * j) : (2 * j + 8)) + khi;   // lanes>=16: khi=8
}

// --- Repack f32 [K,N=128] row-major weights into bf16 WMMA B-fragment order:
//     dword index t = ((kc*8 + nt)*32 + lane)*8 + j
__global__ void repack_kernel(const float* __restrict__ src,
                              unsigned short* __restrict__ dst, int K, int N) {
    int t = blockIdx.x * blockDim.x + threadIdx.x;
    int total = K * N / 2;
    if (t >= total) return;
    int j    = t & 7;
    int lane = (t >> 3) & 31;
    int frag = t >> 8;
    int NT   = N >> 4;
    int nt   = frag % NT;
    int kc   = frag / NT;
    int khi  = (lane >= 16) ? 8 : 0;
    int k0   = kc * 32 + frag_k0(j, khi);
    int n    = nt * 16 + (lane & 15);
    ((unsigned*)dst)[t] = pack_bf16(src[(size_t)k0 * N + n],
                                    src[(size_t)(k0 + 1) * N + n]);
}

__global__ void zero_f32_kernel(float* __restrict__ p, size_t n) {
    size_t i = (size_t)blockIdx.x * blockDim.x + threadIdx.x;
    size_t stride = (size_t)gridDim.x * blockDim.x;
    for (; i < n; i += stride) p[i] = 0.0f;
}

// --- Node linear: hout[N,128](bf16) = (relu?)(x) @ W[128,128] + b, via WMMA.
// 1 wave = 16 rows. 4 waves/block.
__global__ __launch_bounds__(128)
void node_linear_kernel(const float* __restrict__ x, int applyRelu,
                        const unsigned short* __restrict__ wr,
                        const float* __restrict__ bias,
                        unsigned short* __restrict__ hout, int nRows) {
    int lane = threadIdx.x & 31;
    int wv   = threadIdx.x >> 5;
    int tile = blockIdx.x * 4 + wv;
    if (tile * 16 >= nRows) return;

    int mrow = lane & 15;
    int khi  = (lane >= 16) ? 8 : 0;
    int row  = tile * 16 + mrow;
    const float* xr = x + (size_t)row * 128;
    const unsigned* wd = (const unsigned*)wr;

    v8f acc[8] = {};
#pragma unroll
    for (int kc = 0; kc < 4; ++kc) {
        union { v16bf v; unsigned u[8]; } a;
#pragma unroll
        for (int j = 0; j < 8; ++j) {
            int k0 = kc * 32 + frag_k0(j, khi);
            float2 f = *(const float2*)(xr + k0);
            if (applyRelu) { f.x = f.x > 0.f ? f.x : 0.f; f.y = f.y > 0.f ? f.y : 0.f; }
            a.u[j] = pack_bf16(f.x, f.y);
        }
#pragma unroll
        for (int nt = 0; nt < 8; ++nt) {
            union { v16bf v; v4u q[2]; } b;
            const v4u* bp = (const v4u*)(wd + ((size_t)((kc * 8 + nt) * 32 + lane)) * 8);
            b.q[0] = bp[0]; b.q[1] = bp[1];
            acc[nt] = __builtin_amdgcn_wmma_f32_16x16x32_bf16(
                false, a.v, false, b.v, (short)0, acc[nt], false, false);
        }
    }
    // D layout: lane&15 -> column, VGPR r -> row r (+8 for lanes>=16)
#pragma unroll
    for (int nt = 0; nt < 8; ++nt) {
        int n = nt * 16 + (lane & 15);
        float bv = bias[n];
#pragma unroll
        for (int r = 0; r < 8; ++r) {
            int rr = tile * 16 + r + khi;
            hout[(size_t)rr * 128 + n] = f32_to_bf16(acc[nt][r] + bv);
        }
    }
}

// --- Edge MLP + scatter-add:
//   per edge e: z = [h[dst]||h[src]] (256 bf16)
//   m = relu(z @ W1[256,128] + b1) @ W2[128,128] + b2 ; acc[dst] += m
// 1 wave = 16 edges. 4 waves/block. Self loops appended for e >= E.
#define Z_STRIDE 264   // bf16 units (16B-aligned rows, padded)
#define M_STRIDE 136
__global__ __launch_bounds__(128)
void edge_mlp_kernel(const unsigned short* __restrict__ h,
                     const int* __restrict__ ei,
                     const unsigned short* __restrict__ w1r, const float* __restrict__ b1,
                     const unsigned short* __restrict__ w2r, const float* __restrict__ b2,
                     float* __restrict__ acc, int E, int N) {
    __shared__ unsigned short s_z[4][16 * Z_STRIDE];
    __shared__ unsigned short s_m[4][16 * M_STRIDE];
    __shared__ int s_d[4][16];
    __shared__ int s_s[4][16];

    int lane = threadIdx.x & 31;
    int wv   = threadIdx.x >> 5;
    int tile = blockIdx.x * 4 + wv;
    int Etot = E + N;
    int nTiles = (Etot + 15) >> 4;
    if (tile >= nTiles) return;

    if (lane < 16) {
        int e = tile * 16 + lane;
        if (e >= Etot) e = Etot - 1;            // (Etot is a multiple of 16 here)
        int s, d;
        if (e < E) { s = ei[e]; d = ei[E + e]; } else { s = d = e - E; }
        s_d[wv][lane] = d;
        s_s[wv][lane] = s;
    }
    asm volatile("s_wait_dscnt 0" ::: "memory");
    __builtin_amdgcn_wave_barrier();

    // Async gather (memory -> LDS, ASYNCcnt): lane pair (el, half) brings in
    // 64 bf16 of the dst row and 64 bf16 of the src row, 128B each as 8xB128.
    {
        int el = lane >> 1, half = lane & 1;
        int d = s_d[wv][el], s = s_s[wv][el];
        const unsigned short* gd = h + (size_t)d * 128 + half * 64;
        const unsigned short* gs = h + (size_t)s * 128 + half * 64;
        unsigned ld = (unsigned)(size_t)&s_z[wv][el * Z_STRIDE + half * 64];
        unsigned ls = (unsigned)(size_t)&s_z[wv][el * Z_STRIDE + 128 + half * 64];
#pragma unroll
        for (int i = 0; i < 8; ++i) {
            asm volatile("global_load_async_to_lds_b128 %0, %1, off"
                         :: "v"(ld + i * 16), "v"(gd + i * 8) : "memory");
            asm volatile("global_load_async_to_lds_b128 %0, %1, off"
                         :: "v"(ls + i * 16), "v"(gs + i * 8) : "memory");
        }
    }
    asm volatile("s_wait_asynccnt 0" ::: "memory");
    __builtin_amdgcn_wave_barrier();

    int mrow = lane & 15;
    int khi  = (lane >= 16) ? 8 : 0;
    int ncol = lane & 15;
    const unsigned* zrow = (const unsigned*)&s_z[wv][mrow * Z_STRIDE];
    const unsigned* w1d  = (const unsigned*)w1r;
    const unsigned* w2d  = (const unsigned*)w2r;

    // Layer 1: K = 256 (8 chunks), 8 column tiles.
    v8f a1[8] = {};
#pragma unroll
    for (int kc = 0; kc < 8; ++kc) {
        union { v16bf v; unsigned u[8]; } a;
#pragma unroll
        for (int j = 0; j < 8; ++j) {
            int k0 = kc * 32 + frag_k0(j, khi);
            a.u[j] = zrow[k0 >> 1];
        }
#pragma unroll
        for (int nt = 0; nt < 8; ++nt) {
            union { v16bf v; v4u q[2]; } b;
            const v4u* bp = (const v4u*)(w1d + ((size_t)((kc * 8 + nt) * 32 + lane)) * 8);
            b.q[0] = bp[0]; b.q[1] = bp[1];
            a1[nt] = __builtin_amdgcn_wmma_f32_16x16x32_bf16(
                false, a.v, false, b.v, (short)0, a1[nt], false, false);
        }
    }
    // bias + ReLU -> bf16 hidden tile in LDS
#pragma unroll
    for (int nt = 0; nt < 8; ++nt) {
        int n = nt * 16 + ncol;
        float bv = b1[n];
#pragma unroll
        for (int r = 0; r < 8; ++r) {
            float v = a1[nt][r] + bv;
            v = v > 0.f ? v : 0.f;
            s_m[wv][(r + khi) * M_STRIDE + n] = f32_to_bf16(v);
        }
    }
    asm volatile("s_wait_dscnt 0" ::: "memory");
    __builtin_amdgcn_wave_barrier();

    // Layer 2: K = 128 (4 chunks).
    const unsigned* mrp = (const unsigned*)&s_m[wv][mrow * M_STRIDE];
    v8f a2[8] = {};
#pragma unroll
    for (int kc = 0; kc < 4; ++kc) {
        union { v16bf v; unsigned u[8]; } a;
#pragma unroll
        for (int j = 0; j < 8; ++j) {
            int k0 = kc * 32 + frag_k0(j, khi);
            a.u[j] = mrp[k0 >> 1];
        }
#pragma unroll
        for (int nt = 0; nt < 8; ++nt) {
            union { v16bf v; v4u q[2]; } b;
            const v4u* bp = (const v4u*)(w2d + ((size_t)((kc * 8 + nt) * 32 + lane)) * 8);
            b.q[0] = bp[0]; b.q[1] = bp[1];
            a2[nt] = __builtin_amdgcn_wmma_f32_16x16x32_bf16(
                false, a.v, false, b.v, (short)0, a2[nt], false, false);
        }
    }
    // bias + scatter-add to segment-sum accumulator
#pragma unroll
    for (int nt = 0; nt < 8; ++nt) {
        int n = nt * 16 + ncol;
        float bv = b2[n];
#pragma unroll
        for (int r = 0; r < 8; ++r) {
            int erow = r + khi;
            int d = s_d[wv][erow];
            float v = a2[nt][r] + bv;
            __hip_atomic_fetch_add(&acc[(size_t)d * 128 + n], v,
                                   __ATOMIC_RELAXED, __HIP_MEMORY_SCOPE_AGENT);
        }
    }
}

// --- Mean-pool stage 1: sums[g,c] += relu(acc[n,c]); counts[g] += 1
__global__ __launch_bounds__(128)
void pool_kernel(const float* __restrict__ acc, const int* __restrict__ batch,
                 float* __restrict__ sums, float* __restrict__ counts) {
    int nid = blockIdx.x;
    int c   = threadIdx.x;
    int g   = batch[nid];
    float v = acc[(size_t)nid * 128 + c];
    v = v > 0.f ? v : 0.f;
    __hip_atomic_fetch_add(&sums[(size_t)g * 128 + c], v,
                           __ATOMIC_RELAXED, __HIP_MEMORY_SCOPE_AGENT);
    if (c == 0)
        __hip_atomic_fetch_add(&counts[g], 1.0f,
                               __ATOMIC_RELAXED, __HIP_MEMORY_SCOPE_AGENT);
}

// --- Head: out[g,o] = (sums[g,:]/max(cnt,1)) @ out_w[128,64] + out_b
__global__ void head_kernel(const float* __restrict__ sums, const float* __restrict__ counts,
                            const float* __restrict__ ow, const float* __restrict__ ob,
                            float* __restrict__ out) {
    int t = blockIdx.x * blockDim.x + threadIdx.x;
    if (t >= 64 * 64) return;
    int g = t >> 6, o = t & 63;
    float cnt = counts[g];
    float inv = 1.0f / (cnt > 1.0f ? cnt : 1.0f);
    float s = ob[o];
    const float* sg = sums + (size_t)g * 128;
#pragma unroll 4
    for (int c = 0; c < 128; ++c) s += sg[c] * inv * ow[(size_t)c * 64 + o];
    out[t] = s;
}

// ---------------------------------------------------------------------------
extern "C" void kernel_launch(void* const* d_in, const int* in_sizes, int n_in,
                              void* d_out, int out_size, void* d_ws, size_t ws_size,
                              hipStream_t stream) {
    const float* x     = (const float*)d_in[0];
    const int*   ei    = (const int*)d_in[1];
    const int*   batch = (const int*)d_in[3];
    const float* lin_w[3] = {(const float*)d_in[4],  (const float*)d_in[10], (const float*)d_in[16]};
    const float* lin_b[3] = {(const float*)d_in[5],  (const float*)d_in[11], (const float*)d_in[17]};
    const float* w1[3]    = {(const float*)d_in[6],  (const float*)d_in[12], (const float*)d_in[18]};
    const float* b1[3]    = {(const float*)d_in[7],  (const float*)d_in[13], (const float*)d_in[19]};
    const float* w2[3]    = {(const float*)d_in[8],  (const float*)d_in[14], (const float*)d_in[20]};
    const float* b2[3]    = {(const float*)d_in[9],  (const float*)d_in[15], (const float*)d_in[21]};
    const float* out_w = (const float*)d_in[22];
    const float* out_b = (const float*)d_in[23];

    const int N = in_sizes[0] / 128;
    const int E = in_sizes[1] / 2;
    const int G = 64;

    const size_t accB = (size_t)N * 128 * sizeof(float);
    const size_t hB   = (size_t)N * 128 * sizeof(unsigned short);
    const size_t wLayerElems = 16384 + 32768 + 16384;     // lin_r + w1_r + w2_r (bf16)
    const size_t wB   = 3 * wLayerElems * sizeof(unsigned short);
    const size_t sumB = ((size_t)G * 128 + G) * sizeof(float);
    if (ws_size < accB + hB + wB + sumB) return;

    float*          acc    = (float*)d_ws;
    unsigned short* hcur   = (unsigned short*)((char*)d_ws + accB);
    unsigned short* wrbase = (unsigned short*)((char*)d_ws + accB + hB);
    float*          sums   = (float*)((char*)d_ws + accB + hB + wB);
    float*          counts = sums + (size_t)G * 128;

    // 1) Repack all weights into WMMA fragment order (bf16).
    for (int l = 0; l < 3; ++l) {
        unsigned short* linr = wrbase + (size_t)l * wLayerElems;
        unsigned short* w1r  = linr + 16384;
        unsigned short* w2r  = w1r + 32768;
        repack_kernel<<<(8192 + 255) / 256, 256, 0, stream>>>(lin_w[l], linr, 128, 128);
        repack_kernel<<<(16384 + 255) / 256, 256, 0, stream>>>(w1[l], w1r, 256, 128);
        repack_kernel<<<(8192 + 255) / 256, 256, 0, stream>>>(w2[l], w2r, 128, 128);
    }

    const int nodeTiles  = (N + 15) / 16;
    const int nodeBlocks = (nodeTiles + 3) / 4;
    const int edgeTiles  = (E + N + 15) / 16;
    const int edgeBlocks = (edgeTiles + 3) / 4;
    const size_t accElems = (size_t)N * 128;

    // 2) Three MPNN conv layers.
    for (int l = 0; l < 3; ++l) {
        unsigned short* linr = wrbase + (size_t)l * wLayerElems;
        unsigned short* w1r  = linr + 16384;
        unsigned short* w2r  = w1r + 32768;
        const float* xin = (l == 0) ? x : acc;   // layers 2,3 apply relu on load
        node_linear_kernel<<<nodeBlocks, 128, 0, stream>>>(xin, l > 0, linr, lin_b[l], hcur, N);
        zero_f32_kernel<<<2048, 256, 0, stream>>>(acc, accElems);
        edge_mlp_kernel<<<edgeBlocks, 128, 0, stream>>>(hcur, ei, w1r, b1[l], w2r, b2[l],
                                                        acc, E, N);
    }

    // 3) Global mean pool + output projection.
    zero_f32_kernel<<<64, 256, 0, stream>>>(sums, (size_t)G * 128 + G);
    pool_kernel<<<N, 128, 0, stream>>>(acc, batch, sums, counts);
    head_kernel<<<(64 * 64 + 255) / 256, 256, 0, stream>>>(sums, counts, out_w, out_b,
                                                           (float*)d_out);
}
